// model_indRNN_dense_6906307412190
// MI455X (gfx1250) — compile-verified, matrix operand order checked by reading, not run on hardware
//
#include <hip/hip_runtime.h>
#include <hip/hip_bf16.h>
#include <math.h>

// ---------------------------------------------------------------------------
// Types for CDNA5 WMMA (wave32, 16x16x32 bf16 -> f32)
// ---------------------------------------------------------------------------
typedef __attribute__((ext_vector_type(16))) __bf16 v16bf;
typedef __attribute__((ext_vector_type(8)))  __bf16 v8bf;
typedef __attribute__((ext_vector_type(8)))  float  v8f;

// Problem constants (from the reference)
#define BB   24
#define TT   2048
#define IN_D 55
#define HH   512
#define LL   4
#define OUTD 23
#define MROWS (BB * TT)      // 49152
#define K0P  64              // layer-0 K padded 55 -> 64

// ---------------------------------------------------------------------------
// Load an A/B fragment for v_wmma_f32_16x16x32_bf16.
// Per ISA 7.12.2 (16-bit A 16x32): lane l holds row (l&15); elements 0..7 are
// K = 8*(l>>4) .. +7, elements 8..15 are K = 16 + 8*(l>>4) .. +7.
// Caller passes p = rowBase + 8*(lane>>4); we read p[0..7] and p[16..23].
// ---------------------------------------------------------------------------
__device__ __forceinline__ v16bf load_frag16(const __bf16* p) {
  v8bf lo = *(const v8bf*)(p);
  v8bf hi = *(const v8bf*)(p + 16);
  v16bf r;
#pragma unroll
  for (int i = 0; i < 8; ++i) { r[i] = lo[i]; r[i + 8] = hi[i]; }
  return r;
}

// ---------------------------------------------------------------------------
// GEMM: C[M,N] (f32) = A[M,K](bf16) * W[N,K](bf16)^T + bias[N]
// One wave computes a 32(M) x 64(N) macro-tile: 2 A frags x 4 B frags,
// 8 WMMA per 32-wide K step. M%32==0, N%64==0, K%32==0 required.
// ---------------------------------------------------------------------------
__global__ void gemm_bf16_wmma(const __bf16* __restrict__ A,
                               const __bf16* __restrict__ W,
                               const float*  __restrict__ bias,
                               float* __restrict__ C,
                               int M, int N, int K) {
  const int lane   = threadIdx.x & 31;
  const int waveId = (blockIdx.x * blockDim.x + threadIdx.x) >> 5;
  const int nTiles = N >> 6;                    // 64-wide N tiles
  const int m0 = (waveId / nTiles) << 5;        // 32-tall M tiles
  const int n0 = (waveId % nTiles) << 6;
  if (m0 >= M) return;

  const int lm   = lane & 15;
  const int kb8  = (lane >> 4) << 3;            // 0 or 8

  v8f acc[2][4];
#pragma unroll
  for (int i = 0; i < 2; ++i)
#pragma unroll
    for (int j = 0; j < 4; ++j) acc[i][j] = (v8f)(0.0f);

  const __bf16* a0p = A + (size_t)(m0 + lm)      * K + kb8;
  const __bf16* a1p = A + (size_t)(m0 + 16 + lm) * K + kb8;
  const __bf16* wp0 = W + (size_t)(n0 +  0 + lm) * K + kb8;
  const __bf16* wp1 = W + (size_t)(n0 + 16 + lm) * K + kb8;
  const __bf16* wp2 = W + (size_t)(n0 + 32 + lm) * K + kb8;
  const __bf16* wp3 = W + (size_t)(n0 + 48 + lm) * K + kb8;

  for (int k = 0; k < K; k += 32) {
    v16bf a0 = load_frag16(a0p + k);
    v16bf a1 = load_frag16(a1p + k);
    v16bf b0 = load_frag16(wp0 + k);
    v16bf b1 = load_frag16(wp1 + k);
    v16bf b2 = load_frag16(wp2 + k);
    v16bf b3 = load_frag16(wp3 + k);
    acc[0][0] = __builtin_amdgcn_wmma_f32_16x16x32_bf16(false, a0, false, b0, (short)0, acc[0][0], false, false);
    acc[0][1] = __builtin_amdgcn_wmma_f32_16x16x32_bf16(false, a0, false, b1, (short)0, acc[0][1], false, false);
    acc[0][2] = __builtin_amdgcn_wmma_f32_16x16x32_bf16(false, a0, false, b2, (short)0, acc[0][2], false, false);
    acc[0][3] = __builtin_amdgcn_wmma_f32_16x16x32_bf16(false, a0, false, b3, (short)0, acc[0][3], false, false);
    acc[1][0] = __builtin_amdgcn_wmma_f32_16x16x32_bf16(false, a1, false, b0, (short)0, acc[1][0], false, false);
    acc[1][1] = __builtin_amdgcn_wmma_f32_16x16x32_bf16(false, a1, false, b1, (short)0, acc[1][1], false, false);
    acc[1][2] = __builtin_amdgcn_wmma_f32_16x16x32_bf16(false, a1, false, b2, (short)0, acc[1][2], false, false);
    acc[1][3] = __builtin_amdgcn_wmma_f32_16x16x32_bf16(false, a1, false, b3, (short)0, acc[1][3], false, false);
  }

  // C/D layout: VGPR r, lane l -> M = r + 8*(l>>4), N = l&15
  const int halfSel = lane >> 4;
#pragma unroll
  for (int i = 0; i < 2; ++i) {
    const int mb = m0 + i * 16 + 8 * halfSel;
#pragma unroll
    for (int j = 0; j < 4; ++j) {
      const int ncol = n0 + j * 16 + lm;
      const float bv = bias[ncol];
      float* cp = C + (size_t)mb * N + ncol;
#pragma unroll
      for (int r = 0; r < 8; ++r) cp[(size_t)r * N] = acc[i][j][r] + bv;
    }
  }
}

// ---------------------------------------------------------------------------
// IndRNN scan: one thread per (b,h). h_t = relu(wx[b,t,h] + u[h]*h_{t-1}).
// Bias already folded into wx by the GEMM. Writes bf16 activations.
// ---------------------------------------------------------------------------
__global__ void indrnn_scan(const float* __restrict__ wx,
                            const float* __restrict__ u,
                            __bf16* __restrict__ y) {
  const int idx = blockIdx.x * blockDim.x + threadIdx.x;
  if (idx >= BB * HH) return;
  const int b = idx / HH;
  const int h = idx % HH;
  const float uh = u[h];
  const float* p = wx + (size_t)b * TT * HH + h;
  __bf16*      q = y  + (size_t)b * TT * HH + h;
  float hs = 0.0f;
#pragma unroll 8
  for (int t = 0; t < TT; ++t) {
    float v = p[(size_t)t * HH] + uh * hs;
    hs = v > 0.0f ? v : 0.0f;
    q[(size_t)t * HH] = (__bf16)hs;
  }
}

// ---------------------------------------------------------------------------
// Head: one wave per row; lane j < OUTD computes logit j; wave-wide
// shuffle reductions for log_softmax over 23 classes.
// ---------------------------------------------------------------------------
__global__ void head_logsoftmax(const __bf16* __restrict__ Y,
                                const __bf16* __restrict__ Wl,
                                const float*  __restrict__ bl,
                                float* __restrict__ out) {
  const int lane = threadIdx.x & 31;
  const int row  = (blockIdx.x * blockDim.x + threadIdx.x) >> 5;
  if (row >= MROWS) return;

  float logit = -INFINITY;
  if (lane < OUTD) {
    const __bf16* y = Y  + (size_t)row  * HH;
    const __bf16* w = Wl + (size_t)lane * HH;
    float s = 0.0f;
#pragma unroll 4
    for (int h = 0; h < HH; h += 8) {
      v8bf yv = *(const v8bf*)(y + h);
      v8bf wv = *(const v8bf*)(w + h);
#pragma unroll
      for (int i = 0; i < 8; ++i) s += (float)yv[i] * (float)wv[i];
    }
    logit = s + bl[lane];
  }
  float m = logit;
#pragma unroll
  for (int off = 16; off; off >>= 1) m = fmaxf(m, __shfl_xor(m, off, 32));
  float e = (lane < OUTD) ? __expf(logit - m) : 0.0f;
  float se = e;
#pragma unroll
  for (int off = 16; off; off >>= 1) se += __shfl_xor(se, off, 32);
  const float lse = m + __logf(se);
  if (lane < OUTD) out[(size_t)row * OUTD + lane] = logit - lse;
}

// ---------------------------------------------------------------------------
// Convert f32 [R,Kin] -> bf16 [R,Kp], zero-padding columns Kin..Kp-1.
// ---------------------------------------------------------------------------
__global__ void cvt_pad_bf16(const float* __restrict__ in,
                             __bf16* __restrict__ out,
                             int R, int Kin, int Kp) {
  const int i = blockIdx.x * blockDim.x + threadIdx.x;
  if (i >= R * Kp) return;
  const int r = i / Kp;
  const int c = i % Kp;
  out[i] = (c < Kin) ? (__bf16)in[(size_t)r * Kin + c] : (__bf16)0.0f;
}

// ---------------------------------------------------------------------------
// Launch
//   d_in order: x, w0, w_rest, b, u, lin_w, lin_b
// ---------------------------------------------------------------------------
extern "C" void kernel_launch(void* const* d_in, const int* in_sizes, int n_in,
                              void* d_out, int out_size, void* d_ws, size_t ws_size,
                              hipStream_t stream) {
  const float* x     = (const float*)d_in[0];
  const float* w0    = (const float*)d_in[1];
  const float* wrest = (const float*)d_in[2];
  const float* bvec  = (const float*)d_in[3];
  const float* uvec  = (const float*)d_in[4];
  const float* linw  = (const float*)d_in[5];
  const float* linb  = (const float*)d_in[6];
  float* out = (float*)d_out;

  // Workspace layout (all 256B aligned)
  char* ws = (char*)d_ws;
  size_t off = 0;
  auto alloc = [&](size_t bytes) { void* p = ws + off; off = (off + bytes + 255) & ~(size_t)255; return p; };
  __bf16* xb  = (__bf16*)alloc((size_t)MROWS * K0P * sizeof(__bf16));       //  6.3 MB
  __bf16* w0b = (__bf16*)alloc((size_t)HH * K0P * sizeof(__bf16));          //  64 KB
  __bf16* wrb = (__bf16*)alloc((size_t)(LL - 1) * HH * HH * sizeof(__bf16));//  1.5 MB
  __bf16* lwb = (__bf16*)alloc((size_t)OUTD * HH * sizeof(__bf16));         //  23 KB
  float*  wx  = (float*)alloc((size_t)MROWS * HH * sizeof(float));          // 100.7 MB
  __bf16* yb  = (__bf16*)alloc((size_t)MROWS * HH * sizeof(__bf16));        //  50.3 MB
  (void)ws_size; (void)n_in; (void)in_sizes; (void)out_size;

  const int BLK = 256;

  // 1) Convert inputs/weights to bf16 (with K padding where needed)
  {
    int n = MROWS * K0P;
    cvt_pad_bf16<<<(n + BLK - 1) / BLK, BLK, 0, stream>>>(x, xb, MROWS, IN_D, K0P);
    n = HH * K0P;
    cvt_pad_bf16<<<(n + BLK - 1) / BLK, BLK, 0, stream>>>(w0, w0b, HH, IN_D, K0P);
    n = (LL - 1) * HH * HH;
    cvt_pad_bf16<<<(n + BLK - 1) / BLK, BLK, 0, stream>>>(wrest, wrb, (LL - 1) * HH, HH, HH);
    n = OUTD * HH;
    cvt_pad_bf16<<<(n + BLK - 1) / BLK, BLK, 0, stream>>>(linw, lwb, OUTD, HH, HH);
  }

  // GEMM grid: (M/32)*(N/64) waves, 8 waves per 256-thread block
  const int gemmWaves  = (MROWS / 32) * (HH / 64);  // 12288
  const int gemmBlocks = gemmWaves / 8;             // 1536
  const int scanBlocks = (BB * HH + BLK - 1) / BLK; // 48

  // 2) Layer 0: wx = xb @ w0^T + b[0]; then scan
  gemm_bf16_wmma<<<gemmBlocks, BLK, 0, stream>>>(xb, w0b, bvec + 0 * HH, wx, MROWS, HH, K0P);
  indrnn_scan<<<scanBlocks, BLK, 0, stream>>>(wx, uvec + 0 * HH, yb);

  // 3) Layers 1..3
  for (int l = 1; l < LL; ++l) {
    const __bf16* Wl = wrb + (size_t)(l - 1) * HH * HH;
    gemm_bf16_wmma<<<gemmBlocks, BLK, 0, stream>>>(yb, Wl, bvec + (size_t)l * HH, wx, MROWS, HH, HH);
    indrnn_scan<<<scanBlocks, BLK, 0, stream>>>(wx, uvec + (size_t)l * HH, yb);
  }

  // 4) Head: logits + log_softmax (one wave per row)
  {
    const int waves  = MROWS;          // 49152 waves
    const int blocks = waves / 8;      // 6144 blocks of 256 threads
    head_logsoftmax<<<blocks, BLK, 0, stream>>>(yb, lwb, linb, out);
  }
}
